// RationalQuadratic_10582799417482
// MI455X (gfx1250) — matrix operand order repaired and checked by reading, not compiled, not run on hardware
//
#include <hip/hip_runtime.h>
#include <hip/hip_bf16.h>

// ---------------------------------------------------------------------------
// Types for WMMA
// ---------------------------------------------------------------------------
typedef __attribute__((ext_vector_type(16))) __bf16 v16bf;
typedef __attribute__((ext_vector_type(8)))  float  v8f;

union FragAB {
    unsigned int u[8];
    uint4        q[2];   // forces 16B alignment; 2x ds_load_b128 per fragment
    v16bf        v;
};

__device__ __forceinline__ unsigned short f2bf(float f) {
    unsigned int u = __float_as_uint(f);
    unsigned int r = u + 0x7fffu + ((u >> 16) & 1u);   // round-to-nearest-even
    return (unsigned short)(r >> 16);
}

__device__ __forceinline__ float softplus_f(float x) {
    return (x > 20.f) ? x : log1pf(__expf(x));
}

// gfx1250 async global->LDS copy (ASYNCcnt), 16 bytes per lane.
// LDS dest address = low 32 bits of the flat shared pointer (ISA 10.2).
__device__ __forceinline__ void async_b128(unsigned lds_off, const unsigned short* g) {
    asm volatile("global_load_async_to_lds_b128 %0, %1, off"
                 :: "v"(lds_off), "v"(g) : "memory");
}
__device__ __forceinline__ void wait_async0() {
    asm volatile("s_wait_asynccnt 0x0" ::: "memory");
}

// ---------------------------------------------------------------------------
// Problem constants
// ---------------------------------------------------------------------------
#define BATCH   65536
#define D1V     64
#define D2V     64
#define DCV     128
#define DINV    256          // 2*D1 + DC
#define DFFV    1024
#define KBINS   8
#define NPPV    23           // 3*K - 1
#define NOUTV   1472         // NPP * D2
#define MIN_BIN_C   0.1f
#define MIN_DERIV_C 0.001f
#define SHIFT_C     0.5413248546129181f   // log(e - 1)
#define EPS_C       1e-06f

// ---------------------------------------------------------------------------
// Convert + transpose weights: W[K][N] f32  ->  Wt[N][K] bf16
// ---------------------------------------------------------------------------
__global__ void cvt_w_bf16_t(const float* __restrict__ src,
                             unsigned short* __restrict__ dst,
                             int Kd, int Nd, long long total) {
    long long i = (long long)blockIdx.x * blockDim.x + threadIdx.x;
    if (i < total) {
        int k = (int)(i / Nd);
        int n = (int)(i % Nd);
        dst[(long long)n * Kd + k] = f2bf(src[i]);
    }
}

// ---------------------------------------------------------------------------
// Pack h0 = concat(x1, mask1, context) -> bf16  [B, 256]
// ---------------------------------------------------------------------------
__global__ void pack_h0(const float* __restrict__ x1,
                        const float* __restrict__ mask1,
                        const float* __restrict__ ctx,
                        unsigned short* __restrict__ h0) {
    long long i = (long long)blockIdx.x * blockDim.x + threadIdx.x;   // [0, B*256)
    long long b = i >> 8;
    int c = (int)(i & 255);
    float v;
    if (c < 64)       v = x1[b * 64 + c];
    else if (c < 128) v = mask1[b * 64 + (c - 64)];
    else              v = ctx[b * 128 + (c - 128)];
    h0[i] = f2bf(v);
}

// ---------------------------------------------------------------------------
// bf16 WMMA GEMM:  out[M,N] = act(A[M,K] @ Wt[N,K]^T + bias)
//   BM=128, BN=64, BK=32; 256 threads = 8 waves in 4(M) x 2(N) grid;
//   each wave -> 32x32 patch = 2x2 V_WMMA_F32_16X16X32_BF16 tiles.
//   Double-buffered LDS filled by global_load_async_to_lds_b128;
//   XOR swizzle (8-col groups, key (row>>2)&3) keeps both the B128 async
//   writes and the ds_load_b128 fragment reads bank-conflict free.
// ---------------------------------------------------------------------------
template <bool RELU, bool OUT_BF16>
__global__ __launch_bounds__(256)
void gemm_bf16_wmma(const unsigned short* __restrict__ A,    // [M,K] bf16
                    const unsigned short* __restrict__ Wt,   // [N,K] bf16
                    const float* __restrict__ bias,          // [N]
                    void* __restrict__ outp,
                    int Ndim, int Kdim) {
    __shared__ unsigned short As[2][128][32];
    __shared__ unsigned short Bs[2][64][32];

    const int t    = threadIdx.x;
    const int lane = t & 31;
    const int wv   = t >> 5;       // wave 0..7
    const int wm   = wv & 3;       // M quadrant (rows wm*32)
    const int wn   = wv >> 2;      // N half    (cols wn*32)
    const int lo   = lane & 15;
    const int hi   = lane >> 4;

    const long long rowBase = (long long)blockIdx.y * 128;
    const int       colBase = blockIdx.x * 64;
    const unsigned short* Ag = A  + rowBase * Kdim;
    const unsigned short* Bg = Wt + (long long)colBase * Kdim;

    // async-copy chunk assignment (16B per lane per chunk)
    const int arow0 = t >> 2;              // A rows 0..63
    const int arow1 = arow0 + 64;          // A rows 64..127
    const int acol  = (t & 3) * 8;
    const int brow  = t >> 2;              // B rows 0..63
    const int bcol  = (t & 3) * 8;
    const int aks0  = acol ^ (8 * ((arow0 >> 2) & 3));
    const int aks1  = acol ^ (8 * ((arow1 >> 2) & 3));
    const int bks   = bcol ^ (8 * ((brow  >> 2) & 3));

    v8f c[2][2];
    #pragma unroll
    for (int i = 0; i < 2; ++i)
        #pragma unroll
        for (int j = 0; j < 2; ++j)
            c[i][j] = v8f{0.f, 0.f, 0.f, 0.f, 0.f, 0.f, 0.f, 0.f};

    // prologue: fill buffer 0
    async_b128((unsigned)(size_t)&As[0][arow0][aks0], Ag + (long long)arow0 * Kdim + acol);
    async_b128((unsigned)(size_t)&As[0][arow1][aks1], Ag + (long long)arow1 * Kdim + acol);
    async_b128((unsigned)(size_t)&Bs[0][brow][bks],   Bg + (long long)brow  * Kdim + bcol);

    const int nK = Kdim >> 5;
    for (int ks = 0; ks < nK; ++ks) {
        const int cur = ks & 1;
        wait_async0();        // my async writes into buf[cur] complete
        __syncthreads();      // everyone's writes done; prior reads done

        if (ks + 1 < nK) {
            const int nxt = cur ^ 1;
            const int k0  = (ks + 1) << 5;
            async_b128((unsigned)(size_t)&As[nxt][arow0][aks0],
                       Ag + (long long)arow0 * Kdim + k0 + acol);
            async_b128((unsigned)(size_t)&As[nxt][arow1][aks1],
                       Ag + (long long)arow1 * Kdim + k0 + acol);
            async_b128((unsigned)(size_t)&Bs[nxt][brow][bks],
                       Bg + (long long)brow * Kdim + k0 + bcol);
        }

        // ---- fragments: each half is 16 contiguous bytes -> ds_load_b128
        FragAB a[2], b[2];
        #pragma unroll
        for (int tm = 0; tm < 2; ++tm) {
            const int m   = wm * 32 + tm * 16 + lo;
            const int key = 8 * ((m >> 2) & 3);
            const int kb  = hi * 8;
            a[tm].q[0] = *(const uint4*)&As[cur][m][kb ^ key];
            a[tm].q[1] = *(const uint4*)&As[cur][m][(16 + kb) ^ key];
        }
        #pragma unroll
        for (int tn = 0; tn < 2; ++tn) {
            const int n   = wn * 32 + tn * 16 + lo;
            const int key = 8 * ((n >> 2) & 3);
            const int kb2 = hi * 8;
            b[tn].q[0] = *(const uint4*)&Bs[cur][n][kb2 ^ key];
            b[tn].q[1] = *(const uint4*)&Bs[cur][n][(kb2 + 16) ^ key];
        }

        #pragma unroll
        for (int tm = 0; tm < 2; ++tm)
            #pragma unroll
            for (int tn = 0; tn < 2; ++tn)
                c[tm][tn] = __builtin_amdgcn_wmma_f32_16x16x32_bf16(
                    false, a[tm].v, false, b[tn].v,
                    (short)0, c[tm][tn], false, false);
    }

    // ---- epilogue: C/D layout: lane 0-15 -> M=r, lane 16-31 -> M=8+r; N=lo
    unsigned short* outb = (unsigned short*)outp;
    float*          outf = (float*)outp;
    #pragma unroll
    for (int tm = 0; tm < 2; ++tm) {
        #pragma unroll
        for (int tn = 0; tn < 2; ++tn) {
            int gcol = colBase + wn * 32 + tn * 16 + lo;
            float bv = bias[gcol];
            #pragma unroll
            for (int r = 0; r < 8; ++r) {
                long long grow = rowBase + wm * 32 + tm * 16 + hi * 8 + r;
                float v = c[tm][tn][r] + bv;
                if (RELU) v = fmaxf(v, 0.f);
                if (OUT_BF16) outb[grow * Ndim + gcol] = f2bf(v);
                else          outf[grow * Ndim + gcol] = v;
            }
        }
    }
}

// ---------------------------------------------------------------------------
// Rational-quadratic spline: one thread per (b, d2); LDS reduce logdet over d2
// ---------------------------------------------------------------------------
__global__ __launch_bounds__(256)
void rq_spline(const float* __restrict__ params,   // [B, D2, 23]
               const float* __restrict__ x2,       // [B, D2]
               float* __restrict__ zout,           // [B, D2]
               float* __restrict__ ldout) {        // [B]
    __shared__ float red[256];

    const long long b = (long long)blockIdx.x * 4 + (threadIdx.x >> 6);
    const int d = threadIdx.x & 63;
    const float* p = params + (b * 64 + d) * NPPV;
    const float x = x2[b * 64 + d];

    float cw[KBINS + 1], ch[KBINS + 1], dv[KBINS + 1];
    {
        float m = -1e30f;
        #pragma unroll
        for (int i = 0; i < KBINS; ++i) m = fmaxf(m, p[i]);
        float e[KBINS], se = 0.f;
        #pragma unroll
        for (int i = 0; i < KBINS; ++i) { e[i] = __expf(p[i] - m); se += e[i]; }
        float inv = 1.f / se, acc = 0.f;
        cw[0] = -3.f;
        #pragma unroll
        for (int i = 0; i < KBINS; ++i) {
            acc += MIN_BIN_C + (1.f - MIN_BIN_C * KBINS) * (e[i] * inv);
            cw[i + 1] = 6.f * acc - 3.f;
        }
    }
    {
        float m = -1e30f;
        #pragma unroll
        for (int i = 0; i < KBINS; ++i) m = fmaxf(m, p[KBINS + i]);
        float e[KBINS], se = 0.f;
        #pragma unroll
        for (int i = 0; i < KBINS; ++i) { e[i] = __expf(p[KBINS + i] - m); se += e[i]; }
        float inv = 1.f / se, acc = 0.f;
        ch[0] = -3.f;
        #pragma unroll
        for (int i = 0; i < KBINS; ++i) {
            acc += MIN_BIN_C + (1.f - MIN_BIN_C * KBINS) * (e[i] * inv);
            ch[i + 1] = 6.f * acc - 3.f;
        }
    }
    dv[0] = 1.f; dv[KBINS] = 1.f;
    #pragma unroll
    for (int i = 0; i < KBINS - 1; ++i)
        dv[i + 1] = MIN_DERIV_C + softplus_f(p[2 * KBINS + i] + SHIFT_C);

    int idx = 0;
    #pragma unroll
    for (int i = 0; i <= KBINS; ++i) {
        float bump = (i == KBINS) ? (cw[KBINS] + EPS_C) : cw[i];
        idx += (bump <= x) ? 1 : 0;
    }
    idx = min(max(idx - 1, 0), KBINS - 1);

    float x_k  = cw[idx];
    float x_kd = cw[idx + 1] - cw[idx];
    float y_k  = ch[idx];
    float y_kd = ch[idx + 1] - ch[idx];
    float s    = y_kd / x_kd;
    float d0   = dv[idx];
    float d1   = dv[idx + 1];

    float xi   = (x - x_k) / x_kd;
    float xi1m = xi * (1.f - xi);
    float alpha = y_kd * (s * xi * xi + d0 * xi1m);
    float beta  = s + (d1 + d0 - 2.f * s) * xi1m;
    float z_sp  = y_k + alpha / fmaxf(beta, 1e-08f);
    float omx   = 1.f - xi;
    float num   = s * s * (d1 * xi * xi + 2.f * s * xi1m + d0 * omx * omx);
    float ld_sp = __logf(fmaxf(num, 1e-08f)) - 2.f * __logf(fmaxf(beta, 1e-08f));

    bool inside = (-3.f <= x) && (x < 3.f);
    float z  = inside ? z_sp : x;     // affine tail: weight=1, bias=0
    float le = inside ? ld_sp : 0.f;  // log_weight = 0

    zout[b * 64 + d] = z;

    red[threadIdx.x] = le;
    __syncthreads();
    #pragma unroll
    for (int s2 = 32; s2 > 0; s2 >>= 1) {
        if (d < s2) red[threadIdx.x] += red[threadIdx.x + s2];
        __syncthreads();
    }
    if (d == 0) ldout[b] = red[(threadIdx.x >> 6) << 6];
}

// ---------------------------------------------------------------------------
// Host launcher
// ---------------------------------------------------------------------------
extern "C" void kernel_launch(void* const* d_in, const int* in_sizes, int n_in,
                              void* d_out, int out_size, void* d_ws, size_t ws_size,
                              hipStream_t stream) {
    const float* x1    = (const float*)d_in[0];
    const float* x2    = (const float*)d_in[1];
    const float* ctx   = (const float*)d_in[2];
    const float* mask1 = (const float*)d_in[3];
    const float* W0 = (const float*)d_in[4];
    const float* b0 = (const float*)d_in[5];
    const float* W1 = (const float*)d_in[6];
    const float* b1 = (const float*)d_in[7];
    const float* W2 = (const float*)d_in[8];
    const float* b2 = (const float*)d_in[9];
    const float* W3 = (const float*)d_in[10];
    const float* b3 = (const float*)d_in[11];

    char* ws = (char*)d_ws;
    const size_t nW0 = (size_t)DINV * DFFV;      // 262144
    const size_t nW1 = (size_t)DFFV * DFFV;      // 1048576
    const size_t nW3 = (size_t)DFFV * NOUTV;     // 1507328
    size_t off = 0;
    unsigned short* w0t = (unsigned short*)(ws + off); off += nW0 * 2;
    unsigned short* w1t = (unsigned short*)(ws + off); off += nW1 * 2;
    unsigned short* w2t = (unsigned short*)(ws + off); off += nW1 * 2;
    unsigned short* w3t = (unsigned short*)(ws + off); off += nW3 * 2;
    unsigned short* h0  = (unsigned short*)(ws + off); off += (size_t)BATCH * DINV * 2;
    unsigned short* h1  = (unsigned short*)(ws + off); off += (size_t)BATCH * DFFV * 2;
    unsigned short* h2  = (unsigned short*)(ws + off); off += (size_t)BATCH * DFFV * 2;
    unsigned short* h3  = h1;                                    // alias (h1 dead)
    float* params       = (float*)(ws + off);  off += (size_t)BATCH * NOUTV * 4;

    // 1. weights -> bf16, transposed to [N][K]
    cvt_w_bf16_t<<<(int)((nW0 + 255) / 256), 256, 0, stream>>>(W0, w0t, DINV, DFFV, (long long)nW0);
    cvt_w_bf16_t<<<(int)((nW1 + 255) / 256), 256, 0, stream>>>(W1, w1t, DFFV, DFFV, (long long)nW1);
    cvt_w_bf16_t<<<(int)((nW1 + 255) / 256), 256, 0, stream>>>(W2, w2t, DFFV, DFFV, (long long)nW1);
    cvt_w_bf16_t<<<(int)((nW3 + 255) / 256), 256, 0, stream>>>(W3, w3t, DFFV, NOUTV, (long long)nW3);

    // 2. pack h0
    pack_h0<<<(BATCH * DINV) / 256, 256, 0, stream>>>(x1, mask1, ctx, h0);

    // 3..6. MLP
    dim3 blk(256);
    gemm_bf16_wmma<true, true ><<<dim3(DFFV / 64,  BATCH / 128), blk, 0, stream>>>(
        h0, w0t, b0, (void*)h1, DFFV, DINV);
    gemm_bf16_wmma<true, true ><<<dim3(DFFV / 64,  BATCH / 128), blk, 0, stream>>>(
        h1, w1t, b1, (void*)h2, DFFV, DFFV);
    gemm_bf16_wmma<true, true ><<<dim3(DFFV / 64,  BATCH / 128), blk, 0, stream>>>(
        h2, w2t, b2, (void*)h3, DFFV, DFFV);
    gemm_bf16_wmma<false, false><<<dim3(NOUTV / 64, BATCH / 128), blk, 0, stream>>>(
        h3, w3t, b3, (void*)params, NOUTV, DFFV);

    // 7. spline + logdet reduction
    float* zout  = (float*)d_out;                   // [B, 64]
    float* ldout = zout + (size_t)BATCH * D2V;      // [B]
    rq_spline<<<BATCH / 4, 256, 0, stream>>>(params, x2, zout, ldout);
}
